// ProbAttention_51178830299335
// MI455X (gfx1250) — matrix operand order repaired and checked by reading, not compiled, not run on hardware
//
#include <hip/hip_runtime.h>

typedef __attribute__((ext_vector_type(16))) _Float16 v16h;
typedef __attribute__((ext_vector_type(8)))  _Float16 v8h;
typedef __attribute__((ext_vector_type(8)))  float    v8f;

constexpr int NB   = 2;
constexpr int NL   = 4096;
constexpr int ND   = 512;
constexpr int NH   = 8;
constexpr int NE   = 64;
constexpr int NU   = 45;   // top-k selected queries per (b,h)
constexpr int NSK  = 45;   // sampled keys per query
constexpr int UPAD = 48;   // padded to 3 WMMA row-tiles

__device__ __forceinline__ v8f wmma_f16(v16h a, v16h b, v8f c) {
  // D = A(16x32 f16) * B(32x16 f16) + C(16x16 f32)
  return __builtin_amdgcn_wmma_f32_16x16x32_f16(false, a, false, b, (short)0, c,
                                                false, false);
}

// Pack 16 consecutive "logical" floats (given as 4 float4s) into a v16h,
// scaled. Element order matches either operand pattern when the caller picks
// the right float4s.
__device__ __forceinline__ v16h pack16(float4 r0, float4 r1, float4 r2,
                                       float4 r3, float s) {
  v16h a;
  a[0] = (_Float16)(r0.x * s);  a[1] = (_Float16)(r0.y * s);
  a[2] = (_Float16)(r0.z * s);  a[3] = (_Float16)(r0.w * s);
  a[4] = (_Float16)(r1.x * s);  a[5] = (_Float16)(r1.y * s);
  a[6] = (_Float16)(r1.z * s);  a[7] = (_Float16)(r1.w * s);
  a[8] = (_Float16)(r2.x * s);  a[9] = (_Float16)(r2.y * s);
  a[10] = (_Float16)(r2.z * s); a[11] = (_Float16)(r2.w * s);
  a[12] = (_Float16)(r3.x * s); a[13] = (_Float16)(r3.y * s);
  a[14] = (_Float16)(r3.z * s); a[15] = (_Float16)(r3.w * s);
  return a;
}

// ---------------------------------------------------------------------------
// Stage 1: M[b,h,q] = max_s(q.k_s) - sum_s(q.k_s)/L   over 45 sampled keys.
// One wave per query; lanes hold samples {lane, lane+32}.
// ---------------------------------------------------------------------------
__global__ __launch_bounds__(256) void prob_sample_m(
    const float* __restrict__ Q, const float* __restrict__ K,
    const int* __restrict__ idx, float* __restrict__ M) {
  int wid  = (blockIdx.x * blockDim.x + threadIdx.x) >> 5;  // query slot
  int lane = threadIdx.x & 31;
  int qi = wid & (NL - 1);
  int bh = wid >> 12;           // NL == 2^12
  int b = bh >> 3, h = bh & 7;

  const float* qp = Q + ((size_t)b * NL + qi) * ND + h * NE;
  int  i0 = idx[qi * NSK + lane];
  bool v1 = (lane + 32) < NSK;
  int  i1 = v1 ? idx[qi * NSK + lane + 32] : i0;

  const float4* q4 = (const float4*)qp;
  const float4* k0 = (const float4*)(K + ((size_t)b * NL + i0) * ND + h * NE);
  const float4* k1 = (const float4*)(K + ((size_t)b * NL + i1) * ND + h * NE);
  float d0 = 0.f, d1 = 0.f;
#pragma unroll
  for (int i = 0; i < 16; ++i) {
    float4 a = q4[i], c = k0[i], d = k1[i];
    d0 += a.x * c.x + a.y * c.y + a.z * c.z + a.w * c.w;
    d1 += a.x * d.x + a.y * d.y + a.z * d.z + a.w * d.w;
  }
  float mx = fmaxf(d0, v1 ? d1 : -__builtin_inff());
  float sm = d0 + (v1 ? d1 : 0.f);
#pragma unroll
  for (int off = 16; off > 0; off >>= 1) {
    mx = fmaxf(mx, __shfl_xor(mx, off, 32));
    sm += __shfl_xor(sm, off, 32);
  }
  if (lane == 0) M[(size_t)bh * NL + qi] = mx - sm * (1.0f / NL);
}

// ---------------------------------------------------------------------------
// Stage 2: per-(b,h) top-45 of M (iterative argmax; first-index tie break).
// Destroys M (recomputed each launch). Pads slots 45..47 with slot 0's index.
// ---------------------------------------------------------------------------
__global__ __launch_bounds__(256) void prob_topk(float* __restrict__ M,
                                                 int* __restrict__ mtop) {
  int bh = blockIdx.x;
  int tid = threadIdx.x;
  float* Mp = M + (size_t)bh * NL;
  __shared__ float sv[256];
  __shared__ int   si[256];

  for (int it = 0; it < NU; ++it) {
    float best = -__builtin_inff();
    int   bi   = 0x7fffffff;
    for (int j = tid; j < NL; j += 256) {
      float val = Mp[j];
      if (val > best) { best = val; bi = j; }
    }
    sv[tid] = best; si[tid] = bi;
    __syncthreads();
    for (int s = 128; s > 0; s >>= 1) {
      if (tid < s) {
        bool take = (sv[tid + s] > sv[tid]) ||
                    (sv[tid + s] == sv[tid] && si[tid + s] < si[tid]);
        if (take) { sv[tid] = sv[tid + s]; si[tid] = si[tid + s]; }
      }
      __syncthreads();
    }
    if (tid == 0) {
      mtop[bh * UPAD + it] = si[0];
      Mp[si[0]] = -__builtin_inff();
    }
    __syncthreads();
  }
  if (tid == 0) {
    int pad = mtop[bh * UPAD];
    mtop[bh * UPAD + NU]     = pad;
    mtop[bh * UPAD + NU + 1] = pad;
    mtop[bh * UPAD + NU + 2] = pad;
  }
}

// ---------------------------------------------------------------------------
// Stage 3a: vmean[b,d] = mean_l V[b,l,d]
// ---------------------------------------------------------------------------
__global__ __launch_bounds__(256) void prob_vmean(const float* __restrict__ V,
                                                  float* __restrict__ vmean) {
  int d = blockIdx.x * 256 + threadIdx.x;  // 0 .. NB*ND-1
  int b = d / ND, dd = d % ND;
  float s = 0.f;
  for (int l = 0; l < NL; ++l) s += V[((size_t)b * NL + l) * ND + dd];
  vmean[d] = s * (1.0f / NL);
}

// Stage 3b: broadcast vmean into the whole output.
__global__ __launch_bounds__(256) void prob_fill(const float* __restrict__ vmean,
                                                 float* __restrict__ out) {
  size_t i = (size_t)blockIdx.x * 256 + threadIdx.x;  // NB*NL*ND elements
  int b  = (int)(i / ((size_t)NL * ND));
  int dd = (int)(i % ND);
  out[i] = vmean[b * ND + dd];
}

// ---------------------------------------------------------------------------
// Stage 4: flash attention for the selected queries. One wave per
// (b, h, 16-row tile). Online softmax over 4096 keys in 32-key blocks.
// ---------------------------------------------------------------------------
__global__ __launch_bounds__(32) void prob_attn_sel(
    const float* __restrict__ Q, const float* __restrict__ K,
    const float* __restrict__ V, const int* __restrict__ mtop,
    float* __restrict__ out) {
  const int lane = threadIdx.x;
  const int half = lane >> 4;
  const int ln   = lane & 15;
  const int tile = blockIdx.x % 3;
  const int bh   = blockIdx.x / 3;
  const int b = bh >> 3, h = bh & 7;

  __shared__ _Float16 Plds[16][32];    // probs tile [m][n] for C->A transpose
  __shared__ _Float16 VldsT[64][32];   // V block transposed [e][key]

  // ---- A operands: 16 selected query rows, scale 1/sqrt(E) folded in ----
  const int m = ln;  // A-layout row for this lane (same for both halves)
  const int qrow = mtop[bh * UPAD + tile * 16 + m];
  const float4* q4 = (const float4*)(Q + ((size_t)b * NL + qrow) * ND + h * NE);
  const float sc = 0.125f;  // 1/sqrt(64)
  // A element i <-> e = i + 16*half  (+32 for the hi operand)
  v16h Alo = pack16(q4[4 * half], q4[4 * half + 1],
                    q4[4 * half + 2], q4[4 * half + 3], sc);
  v16h Ahi = pack16(q4[8 + 4 * half], q4[9 + 4 * half],
                    q4[10 + 4 * half], q4[11 + 4 * half], sc);

  v8f o0 = {}, o1 = {}, o2 = {}, o3 = {};
  float mrun[8], lrun[8];
#pragma unroll
  for (int r = 0; r < 8; ++r) { mrun[r] = -__builtin_inff(); lrun[r] = 0.f; }

  const float* kbase = K + (size_t)b * NL * ND + h * NE;
  const float* vbase = V + (size_t)b * NL * ND + h * NE;

  for (int kb = 0; kb < NL; kb += 32) {
    // ---- B operands from K: B[e][n] = K[key=kb(+16)+n][e] ----
    // B element j <-> e = j + 16*half (+32 for hi): one contiguous 16-float run
    const float4* ka = (const float4*)(kbase + (size_t)(kb + ln) * ND);
    const float4* kc = (const float4*)(kbase + (size_t)(kb + 16 + ln) * ND);
    v16h B0lo = pack16(ka[4 * half], ka[4 * half + 1], ka[4 * half + 2],
                       ka[4 * half + 3], 1.f);
    v16h B0hi = pack16(ka[8 + 4 * half], ka[9 + 4 * half], ka[10 + 4 * half],
                       ka[11 + 4 * half], 1.f);
    v16h B1lo = pack16(kc[4 * half], kc[4 * half + 1], kc[4 * half + 2],
                       kc[4 * half + 3], 1.f);
    v16h B1hi = pack16(kc[8 + 4 * half], kc[9 + 4 * half], kc[10 + 4 * half],
                       kc[11 + 4 * half], 1.f);

    v8f c0 = {}, c1 = {};
    c0 = wmma_f16(Alo, B0lo, c0);
    c0 = wmma_f16(Ahi, B0hi, c0);
    c1 = wmma_f16(Alo, B1lo, c1);
    c1 = wmma_f16(Ahi, B1hi, c1);

    __syncthreads();  // previous iteration's LDS reads are done

    // ---- stage V block transposed: VldsT[e][key-in-block] ----
    {
      const float4* v4 = (const float4*)(vbase + (size_t)(kb + lane) * ND);
#pragma unroll
      for (int e4 = 0; e4 < 16; ++e4) {
        float4 w = v4[e4];
        VldsT[e4 * 4 + 0][lane] = (_Float16)w.x;
        VldsT[e4 * 4 + 1][lane] = (_Float16)w.y;
        VldsT[e4 * 4 + 2][lane] = (_Float16)w.z;
        VldsT[e4 * 4 + 3][lane] = (_Float16)w.w;
      }
    }

    // ---- online softmax; write P tile to LDS ----
    // C element r <-> row m = r + 8*half, col n = ln (16-lane row groups)
#pragma unroll
    for (int r = 0; r < 8; ++r) {
      float s0 = c0[r], s1 = c1[r];
      float bmax = fmaxf(s0, s1);
#pragma unroll
      for (int off = 1; off < 16; off <<= 1)
        bmax = fmaxf(bmax, __shfl_xor(bmax, off, 32));
      float nm = fmaxf(mrun[r], bmax);
      float alpha = __expf(mrun[r] - nm);
      mrun[r] = nm;
      float p0 = __expf(s0 - nm), p1 = __expf(s1 - nm);
      float ps = p0 + p1;
#pragma unroll
      for (int off = 1; off < 16; off <<= 1) ps += __shfl_xor(ps, off, 32);
      lrun[r] = lrun[r] * alpha + ps;
      o0[r] *= alpha; o1[r] *= alpha; o2[r] *= alpha; o3[r] *= alpha;
      Plds[r + 8 * half][ln]      = (_Float16)p0;
      Plds[r + 8 * half][16 + ln] = (_Float16)p1;
    }
    __syncthreads();

    // ---- A2 = P in A-layout (row m, k = i+16*half) ----
    v16h A2;
    {
      const _Float16* pr = &Plds[m][0];
      union { v16h v; v8h h2[2]; } ua;
      ua.h2[0] = *(const v8h*)(pr + 16 * half);
      ua.h2[1] = *(const v8h*)(pr + 16 * half + 8);
      A2 = ua.v;
    }
    // ---- B2_t[k2][n] = V[kb+k2][t*16+n]: contiguous 32B row of VldsT ----
    v16h Bv0 = *(const v16h*)&VldsT[0  + ln][16 * half];
    v16h Bv1 = *(const v16h*)&VldsT[16 + ln][16 * half];
    v16h Bv2 = *(const v16h*)&VldsT[32 + ln][16 * half];
    v16h Bv3 = *(const v16h*)&VldsT[48 + ln][16 * half];
    o0 = wmma_f16(A2, Bv0, o0);
    o1 = wmma_f16(A2, Bv1, o1);
    o2 = wmma_f16(A2, Bv2, o2);
    o3 = wmma_f16(A2, Bv3, o3);
  }

  // ---- normalize and scatter the 45 valid rows ----
#pragma unroll
  for (int r = 0; r < 8; ++r) {
    int mm = r + 8 * half;
    int u = tile * 16 + mm;
    if (u < NU) {
      int orow = mtop[bh * UPAD + u];
      float inv = 1.0f / lrun[r];
      float* ob = out + ((size_t)b * NL + orow) * ND + h * NE;
      ob[0  + ln] = o0[r] * inv;
      ob[16 + ln] = o1[r] * inv;
      ob[32 + ln] = o2[r] * inv;
      ob[48 + ln] = o3[r] * inv;
    }
  }
}

// ---------------------------------------------------------------------------
extern "C" void kernel_launch(void* const* d_in, const int* in_sizes, int n_in,
                              void* d_out, int out_size, void* d_ws,
                              size_t ws_size, hipStream_t stream) {
  const float* Q  = (const float*)d_in[0];
  const float* K  = (const float*)d_in[1];
  const float* V  = (const float*)d_in[2];
  const int* idx  = (const int*)d_in[3];
  float* out = (float*)d_out;

  // workspace layout
  float* M    = (float*)d_ws;                        // NB*NH*NL floats
  int*   mtop = (int*)(M + (size_t)NB * NH * NL);    // NB*NH*UPAD ints
  float* vmean = (float*)(mtop + NB * NH * UPAD);    // NB*ND floats

  // stage 1: sampled sparsity measure (one wave per query)
  prob_sample_m<<<(NB * NH * NL) / 8, 256, 0, stream>>>(Q, K, idx, M);
  // stage 2: per-(b,h) top-45
  prob_topk<<<NB * NH, 256, 0, stream>>>(M, mtop);
  // stage 3: mean(V) broadcast into out
  prob_vmean<<<(NB * ND) / 256, 256, 0, stream>>>(V, vmean);
  prob_fill<<<(int)(((size_t)NB * NL * ND) / 256), 256, 0, stream>>>(vmean, out);
  // stage 4: WMMA flash attention on selected rows, overwrite those rows
  prob_attn_sel<<<NB * NH * 3, 32, 0, stream>>>(Q, K, V, mtop, out);
}
